// Vagrant_4226247819520
// MI455X (gfx1250) — matrix operand order repaired and checked by reading, not compiled, not run on hardware
//
#include <hip/hip_runtime.h>

// ---------------------------------------------------------------------------
// Types for CDNA5 WMMA (wave32): v16bf A/B operands, v8f accumulator
// ---------------------------------------------------------------------------
typedef __attribute__((ext_vector_type(16))) __bf16 v16bf;
typedef __attribute__((ext_vector_type(8)))  __bf16 v8bf;
typedef __attribute__((ext_vector_type(8)))  float  v8f;

union U16bf { v16bf v; v8bf h[2]; };

static __device__ __forceinline__ unsigned short f2bf(float f) {
    unsigned int u = __float_as_uint(f);
    u = (u + 0x7FFFu + ((u >> 16) & 1u)) >> 16;   // RNE
    return (unsigned short)u;
}
static __device__ __forceinline__ float bf2f(unsigned short h) {
    return __uint_as_float(((unsigned int)h) << 16);
}

// ---------------------------------------------------------------------------
// fp32 -> bf16 conversion (weights / misc)
// ---------------------------------------------------------------------------
__global__ void cvt_bf16_kernel(const float* __restrict__ src,
                                unsigned short* __restrict__ dst, long long n) {
    long long i = (long long)blockIdx.x * blockDim.x + threadIdx.x;
    long long stride = (long long)gridDim.x * blockDim.x;
    for (; i < n; i += stride) dst[i] = f2bf(src[i]);
}

// ---------------------------------------------------------------------------
// is_common flag
// ---------------------------------------------------------------------------
__global__ void init_flag_kernel(int* flag) { *flag = 1; }

__global__ void check_common_kernel(const int* __restrict__ ids, int n, int* flag) {
    int i = blockIdx.x * blockDim.x + threadIdx.x;
    if (i < n && ids[i] >= 1000) atomicAnd(flag, 0);
}

// ---------------------------------------------------------------------------
// Embedding gather -> h_bf (B*S x H) bf16
// ---------------------------------------------------------------------------
__global__ __launch_bounds__(256) void embed_kernel(
    const int* __restrict__ ids, const float* __restrict__ emb,
    const float* __restrict__ semb, const int* __restrict__ flag,
    unsigned short* __restrict__ h_bf) {
    int row = blockIdx.x;                 // 0..2047
    int id = ids[row];
    int common = *flag;
    int cid = id < 0 ? 0 : (id > 999 ? 999 : id);
    const float* src = common ? (semb + (size_t)cid * 1024)
                              : (emb  + (size_t)id  * 1024);
    for (int c = threadIdx.x; c < 1024; c += 256)
        h_bf[(size_t)row * 1024 + c] = f2bf(src[c]);
}

// ---------------------------------------------------------------------------
// Layer scoring + top-3 selection (batch 0 only per reference)
// ---------------------------------------------------------------------------
__global__ __launch_bounds__(256) void select_kernel(
    const unsigned short* __restrict__ h_bf, const float* __restrict__ Wls,
    const float* __restrict__ bls, int* __restrict__ sel_out) {
    __shared__ float hm[1024];
    __shared__ float red[256];
    __shared__ float scores[3];
    int tid = threadIdx.x;
    for (int c = tid; c < 1024; c += 256) {
        float s = 0.f;
        for (int r = 0; r < 1024; ++r) s += bf2f(h_bf[(size_t)r * 1024 + c]);
        hm[c] = s * (1.f / 1024.f);
    }
    __syncthreads();
    for (int li = 0; li < 3; ++li) {
        float p = 0.f;
        for (int c = tid; c < 1024; c += 256) p += hm[c] * Wls[li * 1024 + c];
        red[tid] = p; __syncthreads();
        for (int s = 128; s; s >>= 1) {
            if (tid < s) red[tid] += red[tid + s];
            __syncthreads();
        }
        if (tid == 0) scores[li] = red[0] + bls[li];
        __syncthreads();
    }
    if (tid == 0) {
        float sc[3] = { scores[0], scores[1], scores[2] };
        for (int i = 0; i < 3; ++i) {           // stable top-k (ties -> low idx)
            int best = 0; float bv = -3.4e38f;
            for (int j = 0; j < 3; ++j) if (sc[j] > bv) { bv = sc[j]; best = j; }
            sel_out[i] = best; sc[best] = -3.4e38f;
        }
    }
}

// ---------------------------------------------------------------------------
// Per-batch gate: sigmoid(mean_s(Q[b]) . Wbs + bbs) > 0.5  <=>  logit > 0
// ---------------------------------------------------------------------------
__global__ __launch_bounds__(256) void gate_kernel(
    const unsigned short* __restrict__ Qbf, const float* __restrict__ Wbs,
    const float* __restrict__ bbs, int* __restrict__ gate) {
    int b = blockIdx.x;
    __shared__ float red[256];
    const unsigned short* q = Qbf + (size_t)b * 1024 * 1024;
    float acc = 0.f;
    for (int i = threadIdx.x; i < 1024 * 1024; i += 256)
        acc += bf2f(q[i]) * Wbs[i & 1023];
    red[threadIdx.x] = acc; __syncthreads();
    for (int s = 128; s; s >>= 1) {
        if (threadIdx.x < s) red[threadIdx.x] += red[threadIdx.x + s];
        __syncthreads();
    }
    if (threadIdx.x == 0) {
        float v = red[0] * (1.f / 1024.f) + bbs[0];
        gate[b] = (v > 0.f) ? 1 : 0;
    }
}

// ---------------------------------------------------------------------------
// Elementwise gate*up -> bf16
// ---------------------------------------------------------------------------
__global__ void gu_kernel(const unsigned short* __restrict__ g,
                          const unsigned short* __restrict__ u,
                          unsigned short* __restrict__ out, long long n) {
    long long i = (long long)blockIdx.x * blockDim.x + threadIdx.x;
    long long stride = (long long)gridDim.x * blockDim.x;
    for (; i < n; i += stride) out[i] = f2bf(bf2f(g[i]) * bf2f(u[i]));
}

// ---------------------------------------------------------------------------
// Generic WMMA GEMM: C[M,N] = A[M,K] * B[N,K]^T   (bf16 in, fp32 accum)
// Block tile 128x64, 8 waves; wave w owns rows w*16..w*16+15, 4 col tiles.
// Optional: device-side layer select (B += sel[iter]*stride), bf16/fp32 C,
// per-batch gated writeback (S=1024 rows per batch).
// ---------------------------------------------------------------------------
__global__ __launch_bounds__(256) void gemm_bf16_kernel(
    const unsigned short* __restrict__ A,
    const unsigned short* __restrict__ Ball,
    const int* __restrict__ sel, int iter, long long selStride,
    int M, int N, int K,
    unsigned short* __restrict__ Cbf, float* __restrict__ Cf,
    const int* __restrict__ gate) {
    const unsigned short* B = Ball;
    if (sel) B += (long long)sel[iter] * selStride;

    __shared__ __attribute__((aligned(16))) unsigned short As[128 * 40];
    __shared__ __attribute__((aligned(16))) unsigned short Bs[64 * 40];

    const int tid  = threadIdx.x;
    const int w    = tid >> 5;
    const int lane = tid & 31;
    const int nn   = lane & 15;
    const int hi   = lane >> 4;
    const int m0   = blockIdx.y * 128;
    const int n0   = blockIdx.x * 64;

    const int arow = tid >> 1, ahalf = (tid & 1) * 16;   // A: 128x32, 16 elems/thr
    const int brow = tid >> 2, bq    = (tid & 3) * 8;    // B:  64x32,  8 elems/thr

    v8f acc[4] = { v8f{}, v8f{}, v8f{}, v8f{} };
    const int kbA = hi ? 8 : 0;
    const int kbB = hi ? 16 : 0;

    for (int k0 = 0; k0 < K; k0 += 32) {
        if (k0 + 32 < K) {
            __builtin_prefetch((const void*)(A + (size_t)(m0 + arow) * K + k0 + 32), 0, 1);
            __builtin_prefetch((const void*)(B + (size_t)(n0 + brow) * K + k0 + 32), 0, 1);
        }
        const v8bf* ga = (const v8bf*)(A + (size_t)(m0 + arow) * K + k0 + ahalf);
        *((v8bf*)&As[arow * 40 + ahalf])     = ga[0];
        *((v8bf*)&As[arow * 40 + ahalf + 8]) = ga[1];
        const v8bf* gb = (const v8bf*)(B + (size_t)(n0 + brow) * K + k0 + bq);
        *((v8bf*)&Bs[brow * 40 + bq]) = gb[0];
        __syncthreads();

        U16bf af;
        af.h[0] = *(const v8bf*)&As[(w * 16 + nn) * 40 + kbA];        // K kb..kb+7
        af.h[1] = *(const v8bf*)&As[(w * 16 + nn) * 40 + kbA + 16];   // K kb+16..kb+23
#pragma unroll
        for (int nt = 0; nt < 4; ++nt) {
            U16bf bf;
            bf.h[0] = *(const v8bf*)&Bs[(nt * 16 + nn) * 40 + kbA];
            bf.h[1] = *(const v8bf*)&Bs[(nt * 16 + nn) * 40 + kbA + 16];
            acc[nt] = __builtin_amdgcn_wmma_f32_16x16x32_bf16(
                false, af.v, false, bf.v, (short)0, acc[nt], false, false);
        }
        __syncthreads();
    }

#pragma unroll
    for (int nt = 0; nt < 4; ++nt) {
#pragma unroll
        for (int r = 0; r < 8; ++r) {
            int mrow = m0 + w * 16 + r + hi * 8;
            int ncol = n0 + nt * 16 + nn;
            if (gate && !gate[mrow >> 10]) continue;     // S=1024 rows per batch
            float v = acc[nt][r];
            if (Cf)  Cf [(size_t)mrow * N + ncol] = v;
            if (Cbf) Cbf[(size_t)mrow * N + ncol] = f2bf(v);
        }
    }
}

// ---------------------------------------------------------------------------
// Fused attention: per block = (query tile of 16, head, batch)
//   scores(16x1024) via WMMA -> LDS -> softmax + aw>0.01 threshold -> rowsum
//   attn[b,s,kv*256+g*64+d] = rho * V[b,s,kv*64+d]
// LDS: exactly 64 KB (score rows; first 16 floats reused for rho broadcast)
// ---------------------------------------------------------------------------
__global__ __launch_bounds__(256) void attn_kernel(
    const unsigned short* __restrict__ Qbf,   // (B*S) x 1024
    const unsigned short* __restrict__ Kbf,   // (B*S) x 256
    const unsigned short* __restrict__ Vbf,   // (B*S) x 256
    unsigned short* __restrict__ attn_bf) {   // (B*S) x 1024
    __shared__ float sc[16 * 1024];

    const int qt   = blockIdx.x;              // 0..63
    const int head = blockIdx.y;              // 0..15
    const int b    = blockIdx.z;              // 0..1
    const int kv = head >> 2, g = head & 3;
    const int tid  = threadIdx.x;
    const int w    = tid >> 5;
    const int lane = tid & 31;
    const int nn   = lane & 15;
    const int hi   = lane >> 4;
    const int kbA  = hi ? 8 : 0;
    const int kbB  = hi ? 16 : 0;

    // A fragments: 16 query rows x 64 (two k-steps of 32), shared by all waves
    const unsigned short* qp =
        Qbf + (size_t)(b * 1024 + qt * 16 + nn) * 1024 + kv * 256 + g * 64;
    U16bf a0, a1;
    a0.h[0] = *(const v8bf*)(qp + kbA);
    a0.h[1] = *(const v8bf*)(qp + kbA + 16);
    a1.h[0] = *(const v8bf*)(qp + 32 + kbA);
    a1.h[1] = *(const v8bf*)(qp + 32 + kbA + 16);

    // Phase 1: 64 key tiles; wave w handles tiles w, w+8, ..., w+56
    for (int jj = 0; jj < 8; ++jj) {
        int j = w + jj * 8;
        const unsigned short* kp =
            Kbf + (size_t)(b * 1024 + j * 16 + nn) * 256 + kv * 64;
        U16bf b0, b1;
        b0.h[0] = *(const v8bf*)(kp + kbA);
        b0.h[1] = *(const v8bf*)(kp + kbA + 16);
        b1.h[0] = *(const v8bf*)(kp + 32 + kbA);
        b1.h[1] = *(const v8bf*)(kp + 32 + kbA + 16);
        v8f acc = {};
        acc = __builtin_amdgcn_wmma_f32_16x16x32_bf16(false, a0.v, false, b0.v,
                                                      (short)0, acc, false, false);
        acc = __builtin_amdgcn_wmma_f32_16x16x32_bf16(false, a1.v, false, b1.v,
                                                      (short)0, acc, false, false);
#pragma unroll
        for (int r = 0; r < 8; ++r)
            sc[(r + hi * 8) * 1024 + j * 16 + nn] = acc[r] * 0.125f;  // 1/sqrt(64)
    }
    __syncthreads();

    // Phase 2: softmax + threshold rowsum; wave w owns rows w and w+8
    float rho_reg[2];
    for (int pass = 0; pass < 2; ++pass) {
        int row = w + pass * 8;
        const float* rp = &sc[row * 1024];
        float mx = -3.4e38f;
        for (int i = lane; i < 1024; i += 32) mx = fmaxf(mx, rp[i]);
        for (int d = 16; d; d >>= 1) mx = fmaxf(mx, __shfl_xor(mx, d, 32));
        float D = 0.f;
        for (int i = lane; i < 1024; i += 32) D += __expf(rp[i] - mx);
        for (int d = 16; d; d >>= 1) D += __shfl_xor(D, d, 32);
        float thr = 0.01f * D;                 // aw > 0.01  <=>  e > 0.01*D
        float rs = 0.f;
        for (int i = lane; i < 1024; i += 32) {
            float e = __expf(rp[i] - mx);
            rs += (e > thr) ? e : 0.f;
        }
        for (int d = 16; d; d >>= 1) rs += __shfl_xor(rs, d, 32);
        rho_reg[pass] = rs / D;
    }
    __syncthreads();                           // all score reads done
    if (lane == 0) { sc[w] = rho_reg[0]; sc[w + 8] = rho_reg[1]; }
    __syncthreads();

    // Phase 3: attn rows = rho * V row slice (16 rows x 64 cols)
    for (int i = tid; i < 16 * 64; i += 256) {
        int row = i >> 6, d = i & 63;
        int srow = b * 1024 + qt * 16 + row;
        float v = sc[row] * bf2f(Vbf[(size_t)srow * 256 + kv * 64 + d]);
        attn_bf[(size_t)srow * 1024 + kv * 256 + g * 64 + d] = f2bf(v);
    }
}

// ---------------------------------------------------------------------------
// Host-side orchestration
// ---------------------------------------------------------------------------
extern "C" void kernel_launch(void* const* d_in, const int* in_sizes, int n_in,
                              void* d_out, int out_size, void* d_ws, size_t ws_size,
                              hipStream_t stream) {
    const int*   ids  = (const int*)  d_in[0];
    const float* emb  = (const float*)d_in[1];
    const float* semb = (const float*)d_in[2];
    const float* Wq   = (const float*)d_in[3];
    const float* Wk   = (const float*)d_in[4];
    const float* Wv   = (const float*)d_in[5];
    const float* Wo   = (const float*)d_in[6];
    const float* Wg   = (const float*)d_in[7];
    const float* Wu   = (const float*)d_in[8];
    const float* Wd   = (const float*)d_in[9];
    const float* Wls  = (const float*)d_in[10];
    const float* bls  = (const float*)d_in[11];
    const float* Wbs  = (const float*)d_in[12];
    const float* bbs  = (const float*)d_in[13];
    const float* Wout = (const float*)d_in[14];
    float* out = (float*)d_out;

    char* p = (char*)d_ws;
    auto allocBF = [&](long long elems) -> unsigned short* {
        unsigned short* r = (unsigned short*)p;
        p += ((size_t)elems * 2 + 255) & ~(size_t)255;
        return r;
    };
    // bf16 weight mirrors
    unsigned short* Wq_bf  = allocBF(3LL * 1024 * 1024);
    unsigned short* Wk_bf  = allocBF(3LL * 256  * 1024);
    unsigned short* Wv_bf  = allocBF(3LL * 256  * 1024);
    unsigned short* Wo_bf  = allocBF(3LL * 1024 * 1024);
    unsigned short* Wg_bf  = allocBF(3LL * 4096 * 1024);
    unsigned short* Wu_bf  = allocBF(3LL * 4096 * 1024);
    unsigned short* Wd_bf  = allocBF(3LL * 1024 * 4096);
    unsigned short* Wout_bf= allocBF(2560LL * 1024);
    // activations (bf16)
    unsigned short* h_bf   = allocBF(2048LL * 1024);
    unsigned short* Qbf    = allocBF(2048LL * 1024);
    unsigned short* Kbf    = allocBF(2048LL * 256);
    unsigned short* Vbf    = allocBF(2048LL * 256);
    unsigned short* Abf    = allocBF(2048LL * 1024);   // attn
    unsigned short* Obf    = allocBF(2048LL * 1024);   // attn @ Wo^T
    unsigned short* Gbf    = allocBF(2048LL * 4096);
    unsigned short* Ubf    = allocBF(2048LL * 4096);
    unsigned short* GUbf   = allocBF(2048LL * 4096);
    // control scalars
    int* flag = (int*)p; p += 256;
    int* sel  = (int*)p; p += 256;
    int* gate = (int*)p; p += 256;

    // 1) weights -> bf16
    cvt_bf16_kernel<<<1024, 256, 0, stream>>>(Wq,   Wq_bf,  3LL*1024*1024);
    cvt_bf16_kernel<<<1024, 256, 0, stream>>>(Wk,   Wk_bf,  3LL*256*1024);
    cvt_bf16_kernel<<<1024, 256, 0, stream>>>(Wv,   Wv_bf,  3LL*256*1024);
    cvt_bf16_kernel<<<1024, 256, 0, stream>>>(Wo,   Wo_bf,  3LL*1024*1024);
    cvt_bf16_kernel<<<1024, 256, 0, stream>>>(Wg,   Wg_bf,  3LL*4096*1024);
    cvt_bf16_kernel<<<1024, 256, 0, stream>>>(Wu,   Wu_bf,  3LL*4096*1024);
    cvt_bf16_kernel<<<1024, 256, 0, stream>>>(Wd,   Wd_bf,  3LL*1024*4096);
    cvt_bf16_kernel<<<1024, 256, 0, stream>>>(Wout, Wout_bf, 2560LL*1024);

    // 2) embedding
    init_flag_kernel<<<1, 1, 0, stream>>>(flag);
    check_common_kernel<<<8, 256, 0, stream>>>(ids, 2048, flag);
    embed_kernel<<<2048, 256, 0, stream>>>(ids, emb, semb, flag, h_bf);

    // 3) layer top-k (device side)
    select_kernel<<<1, 256, 0, stream>>>(h_bf, Wls, bls, sel);

    // 4) TOPL=3 selected layers
    for (int it = 0; it < 3; ++it) {
        gemm_bf16_kernel<<<dim3(16, 16), 256, 0, stream>>>(
            h_bf, Wq_bf, sel, it, 1048576LL, 2048, 1024, 1024, Qbf, nullptr, nullptr);
        gate_kernel<<<2, 256, 0, stream>>>(Qbf, Wbs, bbs, gate);
        gemm_bf16_kernel<<<dim3(4, 16), 256, 0, stream>>>(
            h_bf, Wk_bf, sel, it, 262144LL, 2048, 256, 1024, Kbf, nullptr, nullptr);
        gemm_bf16_kernel<<<dim3(4, 16), 256, 0, stream>>>(
            h_bf, Wv_bf, sel, it, 262144LL, 2048, 256, 1024, Vbf, nullptr, nullptr);
        attn_kernel<<<dim3(64, 16, 2), 256, 0, stream>>>(Qbf, Kbf, Vbf, Abf);
        gemm_bf16_kernel<<<dim3(16, 16), 256, 0, stream>>>(
            Abf, Wo_bf, sel, it, 1048576LL, 2048, 1024, 1024, Obf, nullptr, nullptr);
        gemm_bf16_kernel<<<dim3(64, 16), 256, 0, stream>>>(
            Obf, Wg_bf, sel, it, 4194304LL, 2048, 4096, 1024, Gbf, nullptr, nullptr);
        gemm_bf16_kernel<<<dim3(64, 16), 256, 0, stream>>>(
            Obf, Wu_bf, sel, it, 4194304LL, 2048, 4096, 1024, Ubf, nullptr, nullptr);
        gu_kernel<<<4096, 256, 0, stream>>>(Gbf, Ubf, GUbf, 2048LL * 4096);
        gemm_bf16_kernel<<<dim3(16, 16), 256, 0, stream>>>(
            GUbf, Wd_bf, sel, it, 4194304LL, 2048, 1024, 4096, h_bf, nullptr, gate);
    }

    // 5) output projection: out = h @ Wout[:2560]^T  (fp32 result)
    gemm_bf16_kernel<<<dim3(40, 16), 256, 0, stream>>>(
        h_bf, Wout_bf, nullptr, 0, 0LL, 2048, 2560, 1024, nullptr, out, nullptr);
}
